// EncoderLayer_56195352101169
// MI455X (gfx1250) — matrix operand order repaired and checked by reading, not compiled, not run on hardware
//
#include <hip/hip_runtime.h>

// ---------------------------------------------------------------------------
// MI455X (gfx1250) transformer encoder layer, bf16 WMMA pipeline.
// Compute-bound (~275 GFLOP vs ~150MB HBM traffic): all matmuls run through
// v_wmma_f32_16x16x32_bf16 with fp32 accumulation. Tile staging uses the
// Tensor Data Mover (tensor_load_to_lds, TENSORcnt) with double buffering so
// DMA overlaps the WMMA loop.
// ---------------------------------------------------------------------------

typedef __attribute__((ext_vector_type(16))) __bf16 bf16x16;
typedef __attribute__((ext_vector_type(8)))  __bf16 bf16x8;
typedef __attribute__((ext_vector_type(8)))  float  f32x8;
typedef __attribute__((ext_vector_type(4)))  unsigned int u32x4;
typedef __attribute__((ext_vector_type(8)))  int   i32x8;
typedef __attribute__((ext_vector_type(4)))  int   i32x4;

// ---- WMMA fragment loaders (layouts per CDNA5 ISA 7.12.2) ------
__device__ inline bf16x16 load_a_frag(const __bf16* base, int ld, int lane) {
  const int m = lane & 15, ks = (lane >> 4) & 1;
  const __bf16* p = base + (long)m * ld + ks * 8;
  union { bf16x16 v; bf16x8 h[2]; } u;
  u.h[0] = *(const bf16x8*)(p);
  u.h[1] = *(const bf16x8*)(p + 16);
  return u.v;
}

__device__ inline bf16x16 load_b_frag(const __bf16* base, int ld, int lane) {
  const int n = lane & 15, ks = (lane >> 4) & 1;
  const __bf16* p = base + (long)n * ld + ks * 8;
  union { bf16x16 v; bf16x8 h[2]; } u;
  u.h[0] = *(const bf16x8*)(p);
  u.h[1] = *(const bf16x8*)(p + 16);
  return u.v;
}

__device__ inline f32x8 wmma_bf16(bf16x16 a, bf16x16 b, f32x8 c) {
  return __builtin_amdgcn_wmma_f32_16x16x32_bf16(false, a, false, b,
                                                 (short)0, c, false, false);
}

__device__ inline float red_max16(float v) {
#pragma unroll
  for (int m = 1; m < 16; m <<= 1) v = fmaxf(v, __shfl_xor(v, m, 32));
  return v;
}
__device__ inline float red_sum16(float v) {
#pragma unroll
  for (int m = 1; m < 16; m <<= 1) v += __shfl_xor(v, m, 32);
  return v;
}

// ---------------------------------------------------------------------------
// Tensor Data Mover: 2D bf16 tile (tile_k elems/row x tile_rows rows) from a
// row-major global matrix (row stride = stride_elems) into contiguous LDS.
// D# layout per CDNA5 ISA. Tensor dims are set equal to
// the tile dims (tiles are always fully in-bounds here). 6-arg builtin form
// (this toolchain): (g0, g1, g2, g3, g_extra, cpol).
// ---------------------------------------------------------------------------
__device__ inline void tdm_load_2d(const void* gptr, unsigned lds_off,
                                   unsigned tile_k, unsigned tile_rows,
                                   unsigned stride_elems) {
  const unsigned long long ga = (unsigned long long)(size_t)gptr;
  u32x4 g0;
  g0[0] = 1u;                                   // count=1, user descriptor
  g0[1] = lds_off;                              // lds_addr (bytes)
  g0[2] = (unsigned)(ga & 0xffffffffu);         // global_addr[31:0]
  g0[3] = (unsigned)((ga >> 32) & 0x01ffffffu)  // global_addr[56:32]
          | (2u << 30);                         // type=2 ("image")
  i32x8 g1;
  g1[0] = (int)(1u << 16);                      // data_size=1 (2 bytes)
  g1[1] = (int)((tile_k & 0xffffu) << 16);      // tensor_dim0[15:0]
  g1[2] = (int)(((tile_k >> 16) & 0xffffu) |    // tensor_dim0[31:16]
                ((tile_rows & 0xffffu) << 16)); // tensor_dim1[15:0]
  g1[3] = (int)(((tile_rows >> 16) & 0xffffu) | // tensor_dim1[31:16]
                ((tile_k & 0xffffu) << 16));    // tile_dim0
  g1[4] = (int)(tile_rows & 0xffffu);           // tile_dim1 (tile_dim2=0)
  g1[5] = (int)stride_elems;                    // tensor_dim0_stride[31:0]
  g1[6] = 0;                                    // stride[47:32], dim1_stride lo
  g1[7] = 0;                                    // dim1_stride hi
  const i32x4 z4 = {0, 0, 0, 0};
  const i32x8 z8 = {0, 0, 0, 0, 0, 0, 0, 0};
  __builtin_amdgcn_tensor_load_to_lds(g0, g1, z4, z4, z8, 0);
}

// ---------------------------------------------------------------------------
// Generic GEMM: out[M,N] = A[M,K](bf16) * W[K,N] + bias,  W given as Bt[N,K].
// Block: 256 thr (8 waves), tile 128x128, K-step 32. Wave = 64x32 (4x2 tiles).
// TDM double-buffered LDS staging.
// ---------------------------------------------------------------------------
template <int RELU, int OUTBF>
__global__ __launch_bounds__(256) void gemm_kernel(
    const __bf16* __restrict__ A, const __bf16* __restrict__ Bt,
    const float* __restrict__ bias, void* __restrict__ outp, int K, int N) {
  __shared__ __bf16 Ash[2][128 * 32];
  __shared__ __bf16 Bsh[2][128 * 32];
  const int tid = threadIdx.x, lane = tid & 31, wave = tid >> 5;
  const int wm = wave >> 2, wn = wave & 3;
  const long m0 = (long)blockIdx.x * 128, n0 = (long)blockIdx.y * 128;
  const unsigned ldsA = (unsigned)(size_t)(void*)&Ash[0][0];
  const unsigned ldsB = (unsigned)(size_t)(void*)&Bsh[0][0];
  const unsigned bufBytes = 128 * 32 * 2;

  f32x8 acc[4][2] = {};

  const __bf16* Abase = A + m0 * (long)K;
  const __bf16* Bbase = Bt + n0 * (long)K;
  const int nk = K >> 5;

  if (wave == 0) {
    tdm_load_2d(Abase, ldsA, 32, 128, (unsigned)K);
    tdm_load_2d(Bbase, ldsB, 32, 128, (unsigned)K);
  }

  for (int t = 0; t < nk; ++t) {
    const int cur = t & 1, nxt = cur ^ 1;
    if (wave == 0) {
      if (t + 1 < nk) {
        tdm_load_2d(Abase + (t + 1) * 32, ldsA + nxt * bufBytes, 32, 128,
                    (unsigned)K);
        tdm_load_2d(Bbase + (t + 1) * 32, ldsB + nxt * bufBytes, 32, 128,
                    (unsigned)K);
        __builtin_amdgcn_s_wait_tensorcnt(2);  // current tile landed
      } else {
        __builtin_amdgcn_s_wait_tensorcnt(0);
      }
    }
    __syncthreads();  // release all waves onto buffer `cur`

    bf16x16 bfr[2];
#pragma unroll
    for (int j = 0; j < 2; ++j)
      bfr[j] = load_b_frag(&Bsh[cur][(wn * 32 + j * 16) * 32], 32, lane);
#pragma unroll
    for (int i = 0; i < 4; ++i) {
      bf16x16 af = load_a_frag(&Ash[cur][(wm * 64 + i * 16) * 32], 32, lane);
#pragma unroll
      for (int j = 0; j < 2; ++j) acc[i][j] = wmma_bf16(af, bfr[j], acc[i][j]);
    }
    __syncthreads();  // all waves done with `cur` before TDM reuses it
  }

  const int ks = lane >> 4, nl = lane & 15;
#pragma unroll
  for (int j = 0; j < 2; ++j) {
    const long n = n0 + wn * 32 + j * 16 + nl;
    const float bj = bias[n];
#pragma unroll
    for (int i = 0; i < 4; ++i) {
#pragma unroll
      for (int v = 0; v < 8; ++v) {
        const long row = m0 + wm * 64 + i * 16 + v + 8 * ks;
        float val = acc[i][j][v] + bj;
        if (RELU) val = fmaxf(val, 0.0f);
        if (OUTBF) ((__bf16*)outp)[row * N + n] = (__bf16)val;
        else       ((float*)outp)[row * N + n] = val;
      }
    }
  }
}

// ---------------------------------------------------------------------------
// Flash attention: grid (S/128, B*H). 8 waves; wave owns 16 query rows.
// Q,K: [B*S,1024] bf16 (per-(b,h) base + ld=1024). Vt: [B,H,64,2048] bf16.
// K/V tiles staged via TDM, double buffered.
// ---------------------------------------------------------------------------
__global__ __launch_bounds__(256) void attn_kernel(
    const __bf16* __restrict__ Q, const __bf16* __restrict__ Kmat,
    const __bf16* __restrict__ Vt, __bf16* __restrict__ O) {
  __shared__ __bf16 Ksh[2][128 * 64];     // [key][d]
  __shared__ __bf16 Vsh[2][64 * 128];     // [d][key]
  __shared__ __bf16 Psh[8 * 16 * 128];    // per-wave [row][key]
  const int tid = threadIdx.x, lane = tid & 31, wave = tid >> 5;
  const int ks = lane >> 4, nl = lane & 15;
  const int b = blockIdx.y >> 4, h = blockIdx.y & 15;
  const int q0 = blockIdx.x * 128;
  const unsigned ldsK = (unsigned)(size_t)(void*)&Ksh[0][0];
  const unsigned ldsV = (unsigned)(size_t)(void*)&Vsh[0][0];
  const unsigned tileBytes = 128 * 64 * 2;

  const __bf16* Qbase = Q + ((long)(b * 2048 + q0 + wave * 16)) * 1024 + h * 64;
  bf16x16 qf[2];
  qf[0] = load_a_frag(Qbase + 0, 1024, lane);
  qf[1] = load_a_frag(Qbase + 32, 1024, lane);

  f32x8 o[4] = {};
  float m_run[8], l_run[8];
#pragma unroll
  for (int v = 0; v < 8; ++v) { m_run[v] = -1e30f; l_run[v] = 0.0f; }

  const __bf16* Kg = Kmat + ((long)(b * 2048)) * 1024 + h * 64;  // +kt*1024
  const __bf16* Vg = Vt + ((long)((b * 16 + h) * 64)) * 2048;    // +kt
  __bf16* Pw = Psh + wave * 16 * 128;

  if (wave == 0) {
    tdm_load_2d(Kg, ldsK, 64, 128, 1024);    // 128 keys x 64 d
    tdm_load_2d(Vg, ldsV, 128, 64, 2048);    // 64 d x 128 keys
  }

  const int NT = 2048 / 128;
  for (int t = 0; t < NT; ++t) {
    const int cur = t & 1, nxt = cur ^ 1;
    if (wave == 0) {
      if (t + 1 < NT) {
        tdm_load_2d(Kg + (long)(t + 1) * 128 * 1024, ldsK + nxt * tileBytes,
                    64, 128, 1024);
        tdm_load_2d(Vg + (t + 1) * 128, ldsV + nxt * tileBytes, 128, 64, 2048);
        __builtin_amdgcn_s_wait_tensorcnt(2);
      } else {
        __builtin_amdgcn_s_wait_tensorcnt(0);
      }
    }
    __syncthreads();

    f32x8 sc[8];
#pragma unroll
    for (int n = 0; n < 8; ++n) {
      f32x8 s = {};
      s = wmma_bf16(qf[0], load_b_frag(&Ksh[cur][(n * 16) * 64 + 0], 64, lane), s);
      s = wmma_bf16(qf[1], load_b_frag(&Ksh[cur][(n * 16) * 64 + 32], 64, lane), s);
#pragma unroll
      for (int v = 0; v < 8; ++v) s[v] *= 0.125f;  // 1/sqrt(64)
      sc[n] = s;
    }

    // online softmax (row r lives in one 16-lane half at VGPR index v)
    float mnew[8], scale[8];
#pragma unroll
    for (int v = 0; v < 8; ++v) {
      float mx = sc[0][v];
#pragma unroll
      for (int n = 1; n < 8; ++n) mx = fmaxf(mx, sc[n][v]);
      mx = red_max16(mx);
      mnew[v] = fmaxf(m_run[v], mx);
      scale[v] = __expf(m_run[v] - mnew[v]);
    }
#pragma unroll
    for (int n = 0; n < 8; ++n)
#pragma unroll
      for (int v = 0; v < 8; ++v) sc[n][v] = __expf(sc[n][v] - mnew[v]);
#pragma unroll
    for (int v = 0; v < 8; ++v) {
      float sm = 0.0f;
#pragma unroll
      for (int n = 0; n < 8; ++n) sm += sc[n][v];
      sm = red_sum16(sm);
      l_run[v] = l_run[v] * scale[v] + sm;
      m_run[v] = mnew[v];
    }
#pragma unroll
    for (int nd = 0; nd < 4; ++nd)
#pragma unroll
      for (int v = 0; v < 8; ++v) o[nd][v] *= scale[v];

    // P -> LDS (re-fragment as A operand)
#pragma unroll
    for (int n = 0; n < 8; ++n)
#pragma unroll
      for (int v = 0; v < 8; ++v)
        Pw[(v + 8 * ks) * 128 + n * 16 + nl] = (__bf16)sc[n][v];
    __syncthreads();

    bf16x16 pf[4];
#pragma unroll
    for (int c = 0; c < 4; ++c) pf[c] = load_a_frag(Pw + c * 32, 128, lane);
#pragma unroll
    for (int nd = 0; nd < 4; ++nd)
#pragma unroll
      for (int c = 0; c < 4; ++c)
        o[nd] = wmma_bf16(
            pf[c], load_b_frag(&Vsh[cur][(nd * 16) * 128 + c * 32], 128, lane),
            o[nd]);
    __syncthreads();  // done with `cur` before TDM reuses it
  }

  float inv[8];
#pragma unroll
  for (int v = 0; v < 8; ++v) inv[v] = 1.0f / l_run[v];
#pragma unroll
  for (int nd = 0; nd < 4; ++nd)
#pragma unroll
    for (int v = 0; v < 8; ++v) {
      const long row = (long)(b * 2048 + q0 + wave * 16 + v + 8 * ks);
      O[row * 1024 + h * 64 + nd * 16 + nl] = (__bf16)(o[nd][v] * inv[v]);
    }
}

// ---------------------------------------------------------------------------
// Residual + LayerNorm, one block per row of 1024.
// ---------------------------------------------------------------------------
template <bool WRITE_BF16>
__global__ __launch_bounds__(256) void ln_kernel(
    const float* __restrict__ X, const float* __restrict__ Y,
    const float* __restrict__ g, const float* __restrict__ be,
    float* __restrict__ outF, __bf16* __restrict__ outB) {
  const long row = blockIdx.x;
  const int tid = threadIdx.x;
  const float* xr = X + row * 1024;
  const float* yr = Y + row * 1024;
  float r[4], s = 0.0f, s2 = 0.0f;
#pragma unroll
  for (int i = 0; i < 4; ++i) {
    const int c = tid + i * 256;
    const float t = xr[c] + yr[c];
    r[i] = t; s += t; s2 += t * t;
  }
#pragma unroll
  for (int m = 1; m < 32; m <<= 1) {
    s += __shfl_xor(s, m, 32);
    s2 += __shfl_xor(s2, m, 32);
  }
  __shared__ float ss[8], ss2[8];
  __shared__ float mean_s, rstd_s;
  if ((tid & 31) == 0) { ss[tid >> 5] = s; ss2[tid >> 5] = s2; }
  __syncthreads();
  if (tid == 0) {
    float a = 0.0f, b2 = 0.0f;
    for (int w = 0; w < 8; ++w) { a += ss[w]; b2 += ss2[w]; }
    const float mean = a * (1.0f / 1024.0f);
    const float var = b2 * (1.0f / 1024.0f) - mean * mean;
    mean_s = mean;
    rstd_s = rsqrtf(var + 1e-5f);
  }
  __syncthreads();
  const float mean = mean_s, rstd = rstd_s;
#pragma unroll
  for (int i = 0; i < 4; ++i) {
    const int c = tid + i * 256;
    const float val = (r[i] - mean) * rstd * g[c] + be[c];
    outF[row * 1024 + c] = val;
    if (WRITE_BF16) outB[row * 1024 + c] = (__bf16)val;
  }
}

// ---------------------------------------------------------------------------
// Small data-movement kernels.
// ---------------------------------------------------------------------------
__global__ void f32_to_bf16_kernel(const float* __restrict__ in,
                                   __bf16* __restrict__ out, long n) {
  const long i = (long)blockIdx.x * blockDim.x + threadIdx.x;
  if (i < n) out[i] = (__bf16)in[i];
}

// Wt[n*K + k] = (bf16) W[k*N + n]
__global__ void transpose_w_kernel(const float* __restrict__ W,
                                   __bf16* __restrict__ Wt, int K, int N) {
  const long i = (long)blockIdx.x * blockDim.x + threadIdx.x;
  if (i < (long)K * N) {
    const long n = i / K, k = i - n * K;
    Wt[i] = (__bf16)W[k * (long)N + n];
  }
}

// Vt[((b*16+h)*64+d)*2048 + s] = V[(b*2048+s)*1024 + h*64 + d]
__global__ void transpose_v_kernel(const __bf16* __restrict__ V,
                                   __bf16* __restrict__ Vt) {
  const long i = (long)blockIdx.x * blockDim.x + threadIdx.x;
  const int s = (int)(i & 2047);
  const int d = (int)((i >> 11) & 63);
  const int h = (int)((i >> 17) & 15);
  const int b = (int)(i >> 21);
  Vt[i] = V[((long)b * 2048 + s) * 1024 + h * 64 + d];
}

// ---------------------------------------------------------------------------
extern "C" void kernel_launch(void* const* d_in, const int* in_sizes, int n_in,
                              void* d_out, int out_size, void* d_ws,
                              size_t ws_size, hipStream_t stream) {
  (void)in_sizes; (void)n_in; (void)out_size; (void)ws_size;
  const float* x  = (const float*)d_in[0];
  const float* Wq = (const float*)d_in[1];
  const float* bq = (const float*)d_in[2];
  const float* Wk = (const float*)d_in[3];
  const float* bk = (const float*)d_in[4];
  const float* Wv = (const float*)d_in[5];
  const float* bv = (const float*)d_in[6];
  const float* Wo = (const float*)d_in[7];
  const float* bo = (const float*)d_in[8];
  const float* W1 = (const float*)d_in[9];
  const float* b1 = (const float*)d_in[10];
  const float* W2 = (const float*)d_in[11];
  const float* b2 = (const float*)d_in[12];
  const float* g1 = (const float*)d_in[13];
  const float* be1= (const float*)d_in[14];
  const float* g2 = (const float*)d_in[15];
  const float* be2= (const float*)d_in[16];

  char* ws = (char*)d_ws;
  const size_t Mi = 1ull << 20;
  // lifetime-based reuse (peak 168 MiB)
  __bf16* Xbf = (__bf16*)(ws + 0);        // x bf16        [0,16)
  __bf16* Qb  = (__bf16*)(ws + 16 * Mi);  // Q bf16        [16,32)
  __bf16* Kb  = (__bf16*)(ws + 32 * Mi);  // K bf16        [32,48)
  __bf16* Vb  = (__bf16*)(ws + 48 * Mi);  // V bf16        [48,64)
  __bf16* Vtb = (__bf16*)(ws + 64 * Mi);  // V^T bf16      [64,80)
  __bf16* Ob  = (__bf16*)(ws + 0);        // attn O bf16   reuses Xbf
  float*  attn= (float*)(ws + 16 * Mi);   // attn proj f32 reuses Q,K
  float*  hF  = (float*)(ws + 48 * Mi);   // h f32         reuses V,Vt
  __bf16* hB  = (__bf16*)(ws + 0);        // h bf16        reuses Ob
  __bf16* ff1 = (__bf16*)(ws + 80 * Mi);  // relu(hW1) bf16 [80,144)
  float*  ff2 = (float*)(ws + 16 * Mi);   // ff2 f32       reuses attn
  __bf16* WqT = (__bf16*)(ws + 144 * Mi);
  __bf16* WkT = (__bf16*)(ws + 146 * Mi);
  __bf16* WvT = (__bf16*)(ws + 148 * Mi);
  __bf16* WoT = (__bf16*)(ws + 150 * Mi);
  __bf16* W1T = (__bf16*)(ws + 152 * Mi); // 4096x1024
  __bf16* W2T = (__bf16*)(ws + 160 * Mi); // 1024x4096

  const long NTOK = 8192L * 1024;

  f32_to_bf16_kernel<<<(int)((NTOK + 255) / 256), 256, 0, stream>>>(x, Xbf, NTOK);
  transpose_w_kernel<<<(1024 * 1024) / 256, 256, 0, stream>>>(Wq, WqT, 1024, 1024);
  transpose_w_kernel<<<(1024 * 1024) / 256, 256, 0, stream>>>(Wk, WkT, 1024, 1024);
  transpose_w_kernel<<<(1024 * 1024) / 256, 256, 0, stream>>>(Wv, WvT, 1024, 1024);
  transpose_w_kernel<<<(1024 * 1024) / 256, 256, 0, stream>>>(Wo, WoT, 1024, 1024);
  transpose_w_kernel<<<(4096 * 1024) / 256, 256, 0, stream>>>(W1, W1T, 1024, 4096);
  transpose_w_kernel<<<(4096 * 1024) / 256, 256, 0, stream>>>(W2, W2T, 4096, 1024);

  const dim3 g88(64, 8);
  gemm_kernel<0, 1><<<g88, 256, 0, stream>>>(Xbf, WqT, bq, Qb, 1024, 1024);
  gemm_kernel<0, 1><<<g88, 256, 0, stream>>>(Xbf, WkT, bk, Kb, 1024, 1024);
  gemm_kernel<0, 1><<<g88, 256, 0, stream>>>(Xbf, WvT, bv, Vb, 1024, 1024);

  transpose_v_kernel<<<(int)(NTOK / 256), 256, 0, stream>>>(Vb, Vtb);

  attn_kernel<<<dim3(16, 64), 256, 0, stream>>>(Qb, Kb, Vtb, Ob);

  gemm_kernel<0, 0><<<g88, 256, 0, stream>>>(Ob, WoT, bo, attn, 1024, 1024);
  ln_kernel<true><<<8192, 256, 0, stream>>>(x, attn, g1, be1, hF, hB);

  gemm_kernel<1, 1><<<dim3(64, 32), 256, 0, stream>>>(hB, W1T, b1, ff1, 1024, 4096);
  gemm_kernel<0, 0><<<g88, 256, 0, stream>>>(ff1, W2T, b2, ff2, 4096, 1024);

  ln_kernel<false><<<8192, 256, 0, stream>>>(hF, ff2, g2, be2, (float*)d_out, nullptr);
}
